// TriangleAttention_14293651161600
// MI455X (gfx1250) — matrix operand order repaired and checked by reading, not compiled, not run on hardware
//
#include <hip/hip_runtime.h>
#include <hip/hip_bf16.h>
#include <math.h>

#define N     384
#define NN    (N * N)          // 147456
#define CIN   128
#define CHID  32
#define NH    4
#define INFB  1e9f

// padded LDS row strides; multiples of 8 halves / 4 floats (16B) for b128
#define ZST   136              // f16: 272B = 68 dw (%64 = 4)
#define KST   40               // f16:  80B = 20 dw (%64 = 20)
#define VST   392              // f16: 784B = 196 dw (%64 = 4)
#define OST   132              // f32: 528B = 132 dw (%64 = 4)

typedef _Float16 v16h __attribute__((ext_vector_type(16)));
typedef _Float16 h8   __attribute__((ext_vector_type(8)));
typedef float    v8f  __attribute__((ext_vector_type(8)));

__device__ __forceinline__ v8f vzero8() {
    v8f z;
#pragma unroll
    for (int i = 0; i < 8; ++i) z[i] = 0.f;
    return z;
}

__device__ __forceinline__ v16h cat8(h8 lo, h8 hi) {
    return __builtin_shufflevector(lo, hi, 0, 1, 2, 3, 4, 5, 6, 7,
                                   8, 9, 10, 11, 12, 13, 14, 15);
}

__device__ __forceinline__ v8f wmma_f16(v16h a, v16h b, v8f c) {
    return __builtin_amdgcn_wmma_f32_16x16x32_f16(
        false, a, false, b, (short)0, c, false, false);
}

// A-matrix 16x32 f16, row-major (stride ld halves, rows 16B-aligned)
__device__ __forceinline__ v16h load_A(const _Float16* base, int ld) {
    const int lane = threadIdx.x & 31;
    const _Float16* row = base + (lane & 15) * ld + ((lane < 16) ? 0 : 8);
    h8 lo = *(const h8*)(row);
    h8 hi = *(const h8*)(row + 16);
    return cat8(lo, hi);
}

// B-matrix 32x16 from "n-major" buffer: B[k][n] = base[n*ld + k]
__device__ __forceinline__ v16h load_B_rows(const _Float16* base, int ld) {
    const int lane = threadIdx.x & 31;
    const _Float16* row = base + (lane & 15) * ld + ((lane < 16) ? 0 : 8);
    h8 lo = *(const h8*)(row);
    h8 hi = *(const h8*)(row + 16);
    return cat8(lo, hi);
}

__device__ __forceinline__ float fast_sigmoid(float x) {
    return __builtin_amdgcn_rcpf(1.f + __expf(-x));
}

// ---- k_proj helpers -----------------------------------------------------
__device__ __forceinline__ void stage_wT(const float* __restrict__ W,
                                         _Float16* wT) {
    for (int idx = threadIdx.x; idx < CIN * CIN; idx += 256) {
        int kin = idx >> 7, cout = idx & 127;
        wT[cout * ZST + kin] = (_Float16)W[idx];
    }
}

// MODE: 0 = q (scale 1/sqrt(32)), 3 = g (sigmoid), else passthrough
template <int MODE>
__device__ __forceinline__ void proj_tile_gemm(const _Float16* zn,
                                               const _Float16* wT,
                                               _Float16* outsh) {
    const int tid  = threadIdx.x;
    const int wave = tid >> 5;
    const int lane = tid & 31;
    const int n    = lane & 15;
    const int mhi  = (lane < 16) ? 0 : 8;
    for (int tile = wave; tile < 32; tile += 8) {
        const int rt = tile >> 3;
        const int ct = tile & 7;
        v8f acc = vzero8();
#pragma unroll
        for (int kk = 0; kk < 4; ++kk) {
            v16h A  = load_A(&zn[(rt * 16) * ZST + kk * 32], ZST);
            v16h Bm = load_B_rows(&wT[(ct * 16) * ZST + kk * 32], ZST);
            acc = wmma_f16(A, Bm, acc);
        }
#pragma unroll
        for (int r = 0; r < 8; ++r) {
            float x = acc[r];
            if (MODE == 0) x *= 0.17677669529663687f;   // 1/sqrt(32)
            if (MODE == 3) x = fast_sigmoid(x);
            outsh[(rt * 16 + r + mhi) * ZST + ct * 16 + n] = (_Float16)x;
        }
    }
}

__device__ __forceinline__ void copy_out_f16(const _Float16* outsh,
                                             _Float16* __restrict__ dst,
                                             int rowbase) {
    for (int idx = threadIdx.x; idx < 1024; idx += 256) {  // 64*128/8
        int row = idx >> 4, c8 = (idx & 15) * 8;
        h8 v = *(const h8*)&outsh[row * ZST + c8];
        *(h8*)(dst + (size_t)(rowbase + row) * CIN + c8) = v;
    }
}

// -------------------------------------------------------------------------
// Kernel 1: LayerNorm + tri bias + Q/K/V/G projections (f16 out)
// grid: NN/64 blocks, 256 threads (8 waves); 64 rows per block
// -------------------------------------------------------------------------
__global__ __launch_bounds__(256) void k_proj(
    const float* __restrict__ z, const float* __restrict__ ln_w,
    const float* __restrict__ ln_b, const float* __restrict__ w_tri,
    const float* __restrict__ wq, const float* __restrict__ wk,
    const float* __restrict__ wv, const float* __restrict__ wg,
    _Float16* __restrict__ qb, _Float16* __restrict__ kb,
    _Float16* __restrict__ vb, _Float16* __restrict__ gb,
    float* __restrict__ trib) {
    __shared__ __align__(16) _Float16 zn[64 * ZST];       // 17.0 KB
    __shared__ __align__(16) _Float16 wT[CIN * ZST];      // 34.8 KB
    __shared__ __align__(16) _Float16 outsh[64 * ZST];    // 17.0 KB

    const int tid   = threadIdx.x;
    const int rowq  = tid >> 2;
    const int qlane = tid & 3;
    const int rowg  = blockIdx.x * 64 + rowq;

    // ---- LayerNorm: 4 threads per row, 32 elems each ----
    const float* zrow = z + (size_t)rowg * CIN;
    float vals[32];
    float s = 0.f, ss = 0.f;
#pragma unroll
    for (int u = 0; u < 8; ++u) {
        float4 f = *(const float4*)(zrow + qlane * 4 + u * 16);
        vals[u * 4 + 0] = f.x; vals[u * 4 + 1] = f.y;
        vals[u * 4 + 2] = f.z; vals[u * 4 + 3] = f.w;
        s  += f.x + f.y + f.z + f.w;
        ss += f.x * f.x + f.y * f.y + f.z * f.z + f.w * f.w;
    }
    s  += __shfl_xor(s, 1, 32);  s  += __shfl_xor(s, 2, 32);
    ss += __shfl_xor(ss, 1, 32); ss += __shfl_xor(ss, 2, 32);
    const float mu   = s * (1.f / CIN);
    const float var  = ss * (1.f / CIN) - mu * mu;
    const float rstd = rsqrtf(var + 1e-5f);

    float tpart[NH] = {0.f, 0.f, 0.f, 0.f};
#pragma unroll
    for (int u = 0; u < 8; ++u) {
#pragma unroll
        for (int e = 0; e < 4; ++e) {
            int c = qlane * 4 + u * 16 + e;
            float nv = (vals[u * 4 + e] - mu) * rstd * ln_w[c] + ln_b[c];
            zn[rowq * ZST + c] = (_Float16)nv;
#pragma unroll
            for (int h = 0; h < NH; ++h) tpart[h] += nv * w_tri[c * NH + h];
        }
    }
#pragma unroll
    for (int h = 0; h < NH; ++h) {
        tpart[h] += __shfl_xor(tpart[h], 1, 32);
        tpart[h] += __shfl_xor(tpart[h], 2, 32);
    }
    if (qlane == 0) {
#pragma unroll
        for (int h = 0; h < NH; ++h) trib[(size_t)h * NN + rowg] = tpart[h];
    }
    __syncthreads();

    const int rowbase = blockIdx.x * 64;
    // q
    stage_wT(wq, wT); __syncthreads();
    proj_tile_gemm<0>(zn, wT, outsh); __syncthreads();
    copy_out_f16(outsh, qb, rowbase); __syncthreads();
    // k
    stage_wT(wk, wT); __syncthreads();
    proj_tile_gemm<1>(zn, wT, outsh); __syncthreads();
    copy_out_f16(outsh, kb, rowbase); __syncthreads();
    // v
    stage_wT(wv, wT); __syncthreads();
    proj_tile_gemm<2>(zn, wT, outsh); __syncthreads();
    copy_out_f16(outsh, vb, rowbase); __syncthreads();
    // g (sigmoid)
    stage_wT(wg, wT); __syncthreads();
    proj_tile_gemm<3>(zn, wT, outsh); __syncthreads();
    copy_out_f16(outsh, gb, rowbase);
}

// -------------------------------------------------------------------------
// Kernel 2: fused flash attention per (i, h); og (= q-aliased) f16 out
// grid: (384, 4), 256 threads (8 waves); each wave owns 3 q-tiles of 16 rows
// -------------------------------------------------------------------------
__global__ __launch_bounds__(256) void k_attn(
    const float* __restrict__ mask, const float* __restrict__ trib,
    const _Float16* __restrict__ qb, const _Float16* __restrict__ kb,
    const _Float16* __restrict__ vb, const _Float16* __restrict__ gb,
    _Float16* __restrict__ og) {
    __shared__ __align__(16) _Float16 Ksh[N * KST];        // 30.7 KB (K[j][c])
    __shared__ __align__(16) _Float16 Vt[CHID * VST];      // 25.1 KB (V^T[c][j])
    __shared__ __align__(16) float    biasm[N];            // 1.5 KB
    __shared__ __align__(16) _Float16 Psh[8 * 16 * KST];   // 10.2 KB per-wave

    const int i    = blockIdx.x;
    const int h    = blockIdx.y;
    const int tid  = threadIdx.x;
    const int wave = tid >> 5;
    const int lane = tid & 31;
    const int n    = lane & 15;
    const int mhi  = (lane < 16) ? 0 : 8;

    for (int idx = tid; idx < N * CHID; idx += 256) {
        int j = idx >> 5, c = idx & 31;
        size_t src = (size_t)(i * N + j) * CIN + h * CHID + c;
        Ksh[j * KST + c] = kb[src];
        Vt[c * VST + j]  = vb[src];
    }
    for (int j = tid; j < N; j += 256)
        biasm[j] = INFB * (mask[(size_t)i * N + j] - 1.f);
    __syncthreads();

    const float* trih = trib + (size_t)h * NN;
    _Float16*    P    = &Psh[wave * 16 * KST];

    for (int qt = wave; qt < N / 16; qt += 8) {
        const int qbase = qt * 16;
        v16h Aq = load_A(qb + (size_t)(i * N + qbase) * CIN + h * CHID, CIN);

        v8f O0 = vzero8(), O1 = vzero8();
        float mrun[8], lrun[8];
#pragma unroll
        for (int r = 0; r < 8; ++r) { mrun[r] = -3.0e38f; lrun[r] = 0.f; }

        for (int ch = 0; ch < N / 32; ++ch) {
            const int kbase = ch * 32;
            v16h B0 = load_B_rows(&Ksh[kbase * KST], KST);
            v16h B1 = load_B_rows(&Ksh[(kbase + 16) * KST], KST);
            v8f S0 = wmma_f16(Aq, B0, vzero8());
            v8f S1 = wmma_f16(Aq, B1, vzero8());

            float esc[8];
#pragma unroll
            for (int r = 0; r < 8; ++r) {
                int m = r + mhi;
                const float* trow = trih + (size_t)(qbase + m) * N;
                float s0 = S0[r] + biasm[kbase + n]      + trow[kbase + n];
                float s1 = S1[r] + biasm[kbase + 16 + n] + trow[kbase + 16 + n];
                float a = fmaxf(s0, s1);
#pragma unroll
                for (int msk = 1; msk < 16; msk <<= 1)
                    a = fmaxf(a, __shfl_xor(a, msk, 32));
                float mnew = fmaxf(mrun[r], a);
                esc[r] = __expf(mrun[r] - mnew);
                float p0 = __expf(s0 - mnew), p1 = __expf(s1 - mnew);
                float rs = p0 + p1;
#pragma unroll
                for (int msk = 1; msk < 16; msk <<= 1)
                    rs += __shfl_xor(rs, msk, 32);
                lrun[r] = lrun[r] * esc[r] + rs;
                mrun[r] = mnew;
                O0[r] *= esc[r];
                O1[r] *= esc[r];
                P[m * KST + n]      = (_Float16)p0;
                P[m * KST + 16 + n] = (_Float16)p1;
            }
            asm volatile("s_wait_dscnt 0" ::: "memory");  // wave-local LDS RAW
            v16h Ap  = load_A(P, KST);
            v16h Bv0 = load_B_rows(&Vt[0 * VST + kbase], VST);
            v16h Bv1 = load_B_rows(&Vt[16 * VST + kbase], VST);
            O0 = wmma_f16(Ap, Bv0, O0);
            O1 = wmma_f16(Ap, Bv1, O1);
        }

        // finalize: o/l * gate -> stage into per-wave LDS, then b128 stores
#pragma unroll
        for (int r = 0; r < 8; ++r) {
            int m = r + mhi;
            size_t row = (size_t)(i * N + qbase + m) * CIN + h * CHID;
            float inv = __builtin_amdgcn_rcpf(lrun[r]);
            float g0 = (float)gb[row + n];
            float g1 = (float)gb[row + 16 + n];
            P[m * KST + n]      = (_Float16)(O0[r] * inv * g0);
            P[m * KST + 16 + n] = (_Float16)(O1[r] * inv * g1);
        }
        asm volatile("s_wait_dscnt 0" ::: "memory");
#pragma unroll
        for (int u = 0; u < 2; ++u) {   // 16 rows * 32 halves = 64 h8 chunks
            int idx = lane + 32 * u;
            int row = idx >> 2, c8 = (idx & 3) * 8;
            h8 v = *(const h8*)&P[row * KST + c8];
            *(h8*)(og + (size_t)(i * N + qbase + row) * CIN + h * CHID + c8) = v;
        }
    }
}

// -------------------------------------------------------------------------
// Kernel 3: out = og(f16) @ wo -> fp32 (LDS-staged coalesced stores)
// grid: NN/64 blocks, 256 threads
// -------------------------------------------------------------------------
__global__ __launch_bounds__(256) void k_out(
    const _Float16* __restrict__ og, const float* __restrict__ wo,
    float* __restrict__ out) {
    __shared__ __align__(16) _Float16 wT[CIN * ZST];      // 34.8 KB
    __shared__ __align__(16) float    outsh[64 * OST];    // 33.8 KB

    const int tid  = threadIdx.x;
    const int wave = tid >> 5;
    const int lane = tid & 31;
    const int n    = lane & 15;
    const int mhi  = (lane < 16) ? 0 : 8;

    for (int idx = tid; idx < CIN * CIN; idx += 256) {
        int kin = idx >> 7, cout = idx & 127;
        wT[cout * ZST + kin] = (_Float16)wo[idx];
    }
    __syncthreads();

    const int rowbase = blockIdx.x * 64;
    for (int tile = wave; tile < 32; tile += 8) {
        const int rt = tile >> 3;
        const int ct = tile & 7;
        v8f acc = vzero8();
#pragma unroll
        for (int kk = 0; kk < 4; ++kk) {
            v16h A  = load_A(og + (size_t)(rowbase + rt * 16) * CIN + kk * 32, CIN);
            v16h Bm = load_B_rows(&wT[(ct * 16) * ZST + kk * 32], ZST);
            acc = wmma_f16(A, Bm, acc);
        }
#pragma unroll
        for (int r = 0; r < 8; ++r)
            outsh[(rt * 16 + r + mhi) * OST + ct * 16 + n] = acc[r];
    }
    __syncthreads();

    for (int idx = tid; idx < 2048; idx += 256) {  // 64*128/4 float4 chunks
        int row = idx >> 5, c4 = (idx & 31) * 4;
        float4 v = *(const float4*)&outsh[row * OST + c4];
        *(float4*)(out + (size_t)(rowbase + row) * CIN + c4) = v;
    }
}

// -------------------------------------------------------------------------
extern "C" void kernel_launch(void* const* d_in, const int* in_sizes, int n_in,
                              void* d_out, int out_size, void* d_ws, size_t ws_size,
                              hipStream_t stream) {
    const float* z     = (const float*)d_in[0];
    const float* mask  = (const float*)d_in[1];
    const float* ln_w  = (const float*)d_in[2];
    const float* ln_b  = (const float*)d_in[3];
    const float* w_tri = (const float*)d_in[4];
    const float* wq    = (const float*)d_in[5];
    const float* wk    = (const float*)d_in[6];
    const float* wv    = (const float*)d_in[7];
    const float* wg    = (const float*)d_in[8];
    const float* wo    = (const float*)d_in[9];
    float* out = (float*)d_out;

    char* ws = (char*)d_ws;
    const size_t sz16 = (size_t)NN * CIN * sizeof(_Float16);  // 37.75 MB each
    _Float16* qb   = (_Float16*)(ws);
    _Float16* kb   = (_Float16*)(ws + sz16);
    _Float16* vb   = (_Float16*)(ws + 2 * sz16);
    _Float16* gb   = (_Float16*)(ws + 3 * sz16);
    float*    trib = (float*)   (ws + 4 * sz16);              // 2.36 MB
    _Float16* og   = qb;  // safe alias: (i,h,qtile) regions are RAW-exclusive

    k_proj<<<dim3(NN / 64), dim3(256), 0, stream>>>(
        z, ln_w, ln_b, w_tri, wq, wk, wv, wg, qb, kb, vb, gb, trib);
    k_attn<<<dim3(N, NH), dim3(256), 0, stream>>>(
        mask, trib, qb, kb, vb, gb, og);
    k_out<<<dim3(NN / 64), dim3(256), 0, stream>>>(og, wo, out);
}
